// MoE_76836964925535
// MI455X (gfx1250) — matrix operand order repaired and checked by reading, not compiled, not run on hardware
//
#include <hip/hip_runtime.h>
#include <hip/hip_bf16.h>
#include <stdint.h>

#define HIDDEN_DIM   2048
#define NUM_ROUTED_E 24
#define ROUTED_I     512
#define NUM_SHARED_E 2
#define SHARED_I     2048
#define MOE_TOP_K    6
#define NTOK         4096

typedef __attribute__((ext_vector_type(16))) __bf16 v16bf;
typedef __attribute__((ext_vector_type(2)))  __bf16 v2bf;
typedef __attribute__((ext_vector_type(8)))  float  v8f;

#define BM 128
#define BN 128
#define BK 32
#define LDSW 40   // halves per LDS row: BK + 8 pad = 80 bytes (16B aligned rows)

// native f32->bf16 conversions (v_cvt_bf16_f32 / v_cvt_pk_bf16_f32, RNE)
__device__ __forceinline__ unsigned short f2bf(float f) {
    __bf16 h = (__bf16)f;
    return __builtin_bit_cast(unsigned short, h);
}
__device__ __forceinline__ unsigned int pack2(float a, float b) {
    v2bf p;
    p.x = (__bf16)a;
    p.y = (__bf16)b;
    return __builtin_bit_cast(unsigned int, p);
}

union FragAB {
    uint4 u[2];
    v16bf v;
};

// ---------------------------------------------------------------------------
// Kernel 1: gating — scores = sigmoid(x @ gate_w^T), top-6, normalized weights
// one wave (32 lanes) per token, 4 tokens per 128-thread block
// ---------------------------------------------------------------------------
__global__ __launch_bounds__(128)
void gate_topk_kernel(const float* __restrict__ x,      // [N, K]
                      const float* __restrict__ gw,     // [24, K]
                      float* __restrict__ dense_w,      // [N, 24]
                      int K)
{
    const int lane = threadIdx.x & 31;
    const int wv   = threadIdx.x >> 5;
    const int tok  = blockIdx.x * 4 + wv;

    float acc[NUM_ROUTED_E];
    #pragma unroll
    for (int e = 0; e < NUM_ROUTED_E; ++e) acc[e] = 0.0f;

    const float* xr = x + (size_t)tok * K;
    for (int k = lane; k < K; k += 32) {
        float xv = xr[k];
        #pragma unroll
        for (int e = 0; e < NUM_ROUTED_E; ++e)
            acc[e] += xv * gw[e * K + k];
    }
    #pragma unroll
    for (int e = 0; e < NUM_ROUTED_E; ++e) {
        float v = acc[e];
        #pragma unroll
        for (int off = 16; off > 0; off >>= 1) v += __shfl_xor(v, off, 32);
        acc[e] = v;
    }

    if (lane == 0) {
        float s[NUM_ROUTED_E], w[NUM_ROUTED_E];
        bool used[NUM_ROUTED_E];
        #pragma unroll
        for (int e = 0; e < NUM_ROUTED_E; ++e) {
            s[e] = 1.0f / (1.0f + __expf(-acc[e]));
            w[e] = 0.0f;
            used[e] = false;
        }
        int   idx[MOE_TOP_K];
        float val[MOE_TOP_K];
        float tot = 0.0f;
        #pragma unroll
        for (int k = 0; k < MOE_TOP_K; ++k) {
            float best = -1.0f; int bi = 0;
            #pragma unroll
            for (int e = 0; e < NUM_ROUTED_E; ++e)
                if (!used[e] && s[e] > best) { best = s[e]; bi = e; }
            used[bi] = true; idx[k] = bi; val[k] = best; tot += best;
        }
        tot += 1e-9f;
        #pragma unroll
        for (int k = 0; k < MOE_TOP_K; ++k) w[idx[k]] = val[k] / tot;
        #pragma unroll
        for (int e = 0; e < NUM_ROUTED_E; ++e)
            dense_w[(size_t)tok * NUM_ROUTED_E + e] = w[e];
    }
}

// ---------------------------------------------------------------------------
// Kernel 2: fused SwiGLU up-projection, bf16 WMMA:
//   h[e,n,i] = scl(n,e) * silu(X @ W1_e)[n,i] * (X @ W3_e)[n,i]
// block tile 128x128, 8 waves, each wave 32x64 (2x4 fragments of 16x16),
// K-step 32. fp32 operands converted to bf16 (v_cvt_pk_bf16_f32) during LDS
// staging; W tiles stored transposed in LDS so B fragments are contiguous.
// ---------------------------------------------------------------------------
__global__ __launch_bounds__(256, 1)
void swiglu_gemm_kernel(const float* __restrict__ X,          // [N, K]
                        const float* __restrict__ W1,         // [E, K, I]
                        const float* __restrict__ W3,         // [E, K, I]
                        const float* __restrict__ gate_scale, // [N, 24] or null
                        unsigned short* __restrict__ Hout,    // [E, N, I] bf16
                        int N, int K, int I)
{
    __shared__ alignas(16) unsigned short As [BM * LDSW];
    __shared__ alignas(16) unsigned short B1s[BN * LDSW];
    __shared__ alignas(16) unsigned short B3s[BN * LDSW];
    __shared__ float scl[BM];

    const int tid  = threadIdx.x;
    const int lane = tid & 31;
    const int wave = tid >> 5;
    const int wm   = wave & 3;   // 4 M-groups of 32
    const int wn   = wave >> 2;  // 2 N-groups of 64
    const int e    = blockIdx.z;
    const int m0   = blockIdx.x * BM;
    const int n0   = blockIdx.y * BN;

    if (tid < BM)
        scl[tid] = gate_scale ? gate_scale[(size_t)(m0 + tid) * NUM_ROUTED_E + e] : 1.0f;

    v8f zero = {};
    v8f a1[2][4], a3[2][4];
    #pragma unroll
    for (int i = 0; i < 2; ++i)
        #pragma unroll
        for (int j = 0; j < 4; ++j) { a1[i][j] = zero; a3[i][j] = zero; }

    const size_t wbase = (size_t)e * K * I;

    for (int k0 = 0; k0 < K; k0 += BK) {
        __syncthreads();
        // --- stage X tile: 128x32 f32 -> bf16 (row-major, packed cvt) ---
        #pragma unroll
        for (int i = 0; i < 4; ++i) {
            int c   = tid + i * 256;       // 0..1023 float4 chunks
            int row = c >> 3;              // 0..127
            int c4  = c & 7;               // 0..7
            float4 v = *reinterpret_cast<const float4*>(
                X + (size_t)(m0 + row) * K + (k0 + c4 * 4));
            uint2 p = make_uint2(pack2(v.x, v.y), pack2(v.z, v.w));
            *reinterpret_cast<uint2*>(&As[row * LDSW + c4 * 4]) = p;
        }
        // --- stage W1/W3 tiles: 32x128 f32 -> bf16 transposed [n][k] ---
        // process k in pairs: one v_cvt_pk + one aligned b32 LDS store per pair
        #pragma unroll
        for (int i = 0; i < 2; ++i) {
            int c  = tid + i * 256;        // 0..511
            int kk = (c >> 5) * 2;         // even k: 0..30
            int n4 = (c & 31) * 4;         // 0..124
            size_t g0 = wbase + (size_t)(k0 + kk) * I + (n0 + n4);
            float4 a0 = *reinterpret_cast<const float4*>(W1 + g0);
            float4 a1v = *reinterpret_cast<const float4*>(W1 + g0 + I);
            float4 b0 = *reinterpret_cast<const float4*>(W3 + g0);
            float4 b1v = *reinterpret_cast<const float4*>(W3 + g0 + I);
            *reinterpret_cast<unsigned*>(&B1s[(n4 + 0) * LDSW + kk]) = pack2(a0.x, a1v.x);
            *reinterpret_cast<unsigned*>(&B1s[(n4 + 1) * LDSW + kk]) = pack2(a0.y, a1v.y);
            *reinterpret_cast<unsigned*>(&B1s[(n4 + 2) * LDSW + kk]) = pack2(a0.z, a1v.z);
            *reinterpret_cast<unsigned*>(&B1s[(n4 + 3) * LDSW + kk]) = pack2(a0.w, a1v.w);
            *reinterpret_cast<unsigned*>(&B3s[(n4 + 0) * LDSW + kk]) = pack2(b0.x, b1v.x);
            *reinterpret_cast<unsigned*>(&B3s[(n4 + 1) * LDSW + kk]) = pack2(b0.y, b1v.y);
            *reinterpret_cast<unsigned*>(&B3s[(n4 + 2) * LDSW + kk]) = pack2(b0.z, b1v.z);
            *reinterpret_cast<unsigned*>(&B3s[(n4 + 3) * LDSW + kk]) = pack2(b0.w, b1v.w);
        }
        if (k0 + BK < K) {   // prefetch next K-tile (global_prefetch_b8)
            __builtin_prefetch(X  + (size_t)(m0 + (tid >> 1)) * K + (k0 + BK), 0, 1);
            __builtin_prefetch(W1 + wbase + (size_t)(k0 + BK + (tid >> 3)) * I + n0, 0, 1);
            __builtin_prefetch(W3 + wbase + (size_t)(k0 + BK + (tid >> 3)) * I + n0, 0, 1);
        }
        __syncthreads();

        // --- A fragments (16x32 bf16): lanes 0-15 K0..15,
        //     lanes 16-31 K16..31 (matches B)
        FragAB fa[2];
        #pragma unroll
        for (int i = 0; i < 2; ++i) {
            int m    = wm * 32 + i * 16 + (lane & 15);
            int koff = (lane < 16) ? 0 : 16;
            const unsigned short* p = &As[m * LDSW];
            fa[i].u[0] = *reinterpret_cast<const uint4*>(p + koff);
            fa[i].u[1] = *reinterpret_cast<const uint4*>(p + 8 + koff);
        }
        // --- B fragments (32x16 bf16): lanes 0-15 K0..15, lanes 16-31 K16..31,
        //     contiguous in transposed LDS layout
        #pragma unroll
        for (int j = 0; j < 4; ++j) {
            int n  = wn * 64 + j * 16 + (lane & 15);
            int kb = (lane < 16) ? 0 : 16;
            FragAB fb1, fb3;
            const unsigned short* p1 = &B1s[n * LDSW + kb];
            fb1.u[0] = *reinterpret_cast<const uint4*>(p1);
            fb1.u[1] = *reinterpret_cast<const uint4*>(p1 + 8);
            const unsigned short* p3 = &B3s[n * LDSW + kb];
            fb3.u[0] = *reinterpret_cast<const uint4*>(p3);
            fb3.u[1] = *reinterpret_cast<const uint4*>(p3 + 8);
            #pragma unroll
            for (int i = 0; i < 2; ++i) {
                a1[i][j] = __builtin_amdgcn_wmma_f32_16x16x32_bf16(
                    false, fa[i].v, false, fb1.v, (short)0, a1[i][j], false, false);
                a3[i][j] = __builtin_amdgcn_wmma_f32_16x16x32_bf16(
                    false, fa[i].v, false, fb3.v, (short)0, a3[i][j], false, false);
            }
        }
    }

    // epilogue: h = silu(g) * u * scale, store bf16
    #pragma unroll
    for (int i = 0; i < 2; ++i) {
        #pragma unroll
        for (int j = 0; j < 4; ++j) {
            int nl = wn * 64 + j * 16 + (lane & 15);
            #pragma unroll
            for (int r = 0; r < 8; ++r) {
                int   ml = wm * 32 + i * 16 + r + ((lane < 16) ? 0 : 8);
                float g  = a1[i][j][r];
                float hv = (g / (1.0f + __expf(-g))) * a3[i][j][r] * scl[ml];
                Hout[(size_t)e * N * I + (size_t)(m0 + ml) * I + (n0 + nl)] = f2bf(hv);
            }
        }
    }
}

// ---------------------------------------------------------------------------
// Kernel 3: combine / down-projection with in-register expert reduction:
//   out[n,h] (+)= sum_e h[e,n,:] @ W2[e,:,h]
// A-tile (bf16 h) staged via GLOBAL_LOAD_ASYNC_TO_LDS_B128 (ASYNCcnt path).
// ---------------------------------------------------------------------------
__global__ __launch_bounds__(256, 1)
void combine_gemm_kernel(const unsigned short* __restrict__ Hin, // [E, N, I] bf16
                         const float* __restrict__ W2,           // [E, I, H]
                         float* __restrict__ out,                // [N, H]
                         int N, int I, int H, int E, int accumulate)
{
    __shared__ alignas(16) unsigned short As[BM * LDSW];
    __shared__ alignas(16) unsigned short Bs[BN * LDSW];

    const int tid  = threadIdx.x;
    const int lane = tid & 31;
    const int wave = tid >> 5;
    const int wm   = wave & 3;
    const int wn   = wave >> 2;
    const int m0   = blockIdx.x * BM;
    const int h0   = blockIdx.y * BN;

    v8f zero = {};
    v8f acc[2][4];
    #pragma unroll
    for (int i = 0; i < 2; ++i)
        #pragma unroll
        for (int j = 0; j < 4; ++j) acc[i][j] = zero;

    for (int e = 0; e < E; ++e) {
        const size_t hbase = (size_t)e * N * I;
        const size_t wbase = (size_t)e * I * H;
        for (int k0 = 0; k0 < I; k0 += BK) {
            __syncthreads();
            // --- stage h tile (bf16, 128x32): direct global->LDS async copy ---
            #pragma unroll
            for (int i = 0; i < 2; ++i) {
                int c   = tid + i * 256;   // 0..511 b128 chunks
                int row = c >> 2;          // 0..127
                int q   = c & 3;           // 0..3
                const unsigned short* gp =
                    Hin + hbase + (size_t)(m0 + row) * I + (k0 + q * 8);
                unsigned lds = (unsigned)(uintptr_t)&As[row * LDSW + q * 8];
                asm volatile("global_load_async_to_lds_b128 %0, %1, off"
                             :: "v"(lds), "v"(gp)
                             : "memory");
            }
            // --- stage W2 tile: 32x128 f32 -> bf16 transposed [h][k], k-pairs ---
            #pragma unroll
            for (int i = 0; i < 2; ++i) {
                int c  = tid + i * 256;    // 0..511
                int kk = (c >> 5) * 2;     // even k: 0..30
                int n4 = (c & 31) * 4;     // 0..124
                size_t g0 = wbase + (size_t)(k0 + kk) * H + (h0 + n4);
                float4 a0 = *reinterpret_cast<const float4*>(W2 + g0);
                float4 a1v = *reinterpret_cast<const float4*>(W2 + g0 + H);
                *reinterpret_cast<unsigned*>(&Bs[(n4 + 0) * LDSW + kk]) = pack2(a0.x, a1v.x);
                *reinterpret_cast<unsigned*>(&Bs[(n4 + 1) * LDSW + kk]) = pack2(a0.y, a1v.y);
                *reinterpret_cast<unsigned*>(&Bs[(n4 + 2) * LDSW + kk]) = pack2(a0.z, a1v.z);
                *reinterpret_cast<unsigned*>(&Bs[(n4 + 3) * LDSW + kk]) = pack2(a0.w, a1v.w);
            }
            if (k0 + BK < I)
                __builtin_prefetch(W2 + wbase + (size_t)(k0 + BK + (tid >> 3)) * H + h0, 0, 1);
            // wait for this wave's async LDS writes before the barrier
            asm volatile("s_wait_asynccnt 0x0" ::: "memory");
            __syncthreads();

            FragAB fa[2];
            #pragma unroll
            for (int i = 0; i < 2; ++i) {
                int m    = wm * 32 + i * 16 + (lane & 15);
                int koff = (lane < 16) ? 0 : 16;
                const unsigned short* p = &As[m * LDSW];
                fa[i].u[0] = *reinterpret_cast<const uint4*>(p + koff);
                fa[i].u[1] = *reinterpret_cast<const uint4*>(p + 8 + koff);
            }
            #pragma unroll
            for (int j = 0; j < 4; ++j) {
                int n  = wn * 64 + j * 16 + (lane & 15);
                int kb = (lane < 16) ? 0 : 16;
                FragAB fb;
                const unsigned short* p = &Bs[n * LDSW + kb];
                fb.u[0] = *reinterpret_cast<const uint4*>(p);
                fb.u[1] = *reinterpret_cast<const uint4*>(p + 8);
                #pragma unroll
                for (int i = 0; i < 2; ++i)
                    acc[i][j] = __builtin_amdgcn_wmma_f32_16x16x32_bf16(
                        false, fa[i].v, false, fb.v, (short)0, acc[i][j], false, false);
            }
        }
    }

    #pragma unroll
    for (int i = 0; i < 2; ++i) {
        #pragma unroll
        for (int j = 0; j < 4; ++j) {
            int nl = wn * 64 + j * 16 + (lane & 15);
            #pragma unroll
            for (int r = 0; r < 8; ++r) {
                int    ml  = wm * 32 + i * 16 + r + ((lane < 16) ? 0 : 8);
                size_t idx = (size_t)(m0 + ml) * H + (h0 + nl);
                float  v   = acc[i][j][r];
                if (accumulate) out[idx] += v;
                else            out[idx]  = v;
            }
        }
    }
}

__global__ void write_aux_kernel(float* __restrict__ out, size_t off) {
    if (threadIdx.x == 0 && blockIdx.x == 0) out[off] = 0.0f;
}

// ---------------------------------------------------------------------------
extern "C" void kernel_launch(void* const* d_in, const int* in_sizes, int n_in,
                              void* d_out, int out_size, void* d_ws, size_t ws_size,
                              hipStream_t stream)
{
    (void)in_sizes; (void)n_in; (void)out_size; (void)ws_size;

    const float* x         = (const float*)d_in[0];  // [2,2048,2048] -> [4096,2048]
    const float* gate_w    = (const float*)d_in[1];  // [24,2048]
    const float* shared_w1 = (const float*)d_in[2];  // [2,2048,2048]
    const float* shared_w2 = (const float*)d_in[3];  // [2,2048,2048]
    const float* shared_w3 = (const float*)d_in[4];  // [2,2048,2048]
    const float* routed_w1 = (const float*)d_in[5];  // [24,2048,512]
    const float* routed_w2 = (const float*)d_in[6];  // [24,512,2048]
    const float* routed_w3 = (const float*)d_in[7];  // [24,2048,512]
    float* out = (float*)d_out;                      // [4096,2048] + aux scalar

    // workspace layout
    float*          dense_w = (float*)d_ws;                               // 4096*24 f32 = 384 KB
    unsigned short* h_sh    = (unsigned short*)((char*)d_ws + 0x60000);   // [2,4096,2048] bf16 = 32 MB
    unsigned short* h_rt    = h_sh + (size_t)NUM_SHARED_E * NTOK * SHARED_I; // [24,4096,512] bf16 = 96 MB

    // 1) gating
    gate_topk_kernel<<<NTOK / 4, 128, 0, stream>>>(x, gate_w, dense_w, HIDDEN_DIM);

    // 2) shared-expert SwiGLU up-projection
    swiglu_gemm_kernel<<<dim3(NTOK / BM, SHARED_I / BN, NUM_SHARED_E), 256, 0, stream>>>(
        x, shared_w1, shared_w3, nullptr, h_sh, NTOK, HIDDEN_DIM, SHARED_I);

    // 3) routed-expert SwiGLU up-projection (routing weight folded into h)
    swiglu_gemm_kernel<<<dim3(NTOK / BM, ROUTED_I / BN, NUM_ROUTED_E), 256, 0, stream>>>(
        x, routed_w1, routed_w3, dense_w, h_rt, NTOK, HIDDEN_DIM, ROUTED_I);

    // 4) combine: shared experts (overwrite), then routed experts (accumulate)
    combine_gemm_kernel<<<dim3(NTOK / BM, HIDDEN_DIM / BN), 256, 0, stream>>>(
        h_sh, shared_w2, out, NTOK, SHARED_I, HIDDEN_DIM, NUM_SHARED_E, 0);
    combine_gemm_kernel<<<dim3(NTOK / BM, HIDDEN_DIM / BN), 256, 0, stream>>>(
        h_rt, routed_w2, out, NTOK, ROUTED_I, HIDDEN_DIM, NUM_ROUTED_E, 1);

    // 5) aux loss scalar (eval mode -> 0)
    write_aux_kernel<<<1, 1, 0, stream>>>(out, (size_t)NTOK * HIDDEN_DIM);
}
